// SetCriterion_65773129171134
// MI455X (gfx1250) — compile-verified, matrix operand order checked by reading, not run on hardware
//
#include <hip/hip_runtime.h>
#include <hip/hip_bf16.h>

// Problem constants from the reference
static constexpr int Bn = 4;
static constexpr int Cc = 200;
static constexpr int Gg = 32;
static constexpr int Nn = 32256;              // sum of TEMPORAL_LENGTHS
static constexpr int NTILES = Nn / 16;        // 2016 anchor tiles of 16
static constexpr int CTILES = (Cc + 15) / 16; // 13 class tiles

typedef __attribute__((ext_vector_type(16))) _Float16 v16h;
typedef __attribute__((ext_vector_type(8)))  float    v8f;

__device__ __forceinline__ int lvl_of(int n) {
    return (n < 16384) ? 0 : (n < 24576) ? 1 : (n < 28672) ? 2
         : (n < 30720) ? 3 : (n < 31744) ? 4 : 5;
}

// acc layout in workspace: [0]=focal sum, [1]=seg sum, [2]=iou sum, [3]=num_pos
__global__ void k_init(float* acc) {
    if (threadIdx.x < 4) acc[threadIdx.x] = 0.0f;
}

// One thread per (b,n): build 32-bit positive mask over GTs, accumulate
// smooth-L1 + IoU losses and the positive count. Blocks are b-uniform
// (Nn % 256 == 0), so GT segments are staged in LDS once.
__global__ void k_pairs(const float* __restrict__ pseg,
                        const float* __restrict__ grids,
                        const float* __restrict__ fps,
                        const float* __restrict__ gtseg,
                        unsigned int* __restrict__ pos_mask,
                        float* __restrict__ acc) {
    int idx = blockIdx.x * blockDim.x + threadIdx.x;
    int b = (blockIdx.x * blockDim.x) / Nn;        // uniform per block
    int n = idx - b * Nn;

    __shared__ float sgt[Gg * 2];
    if (threadIdx.x < Gg * 2) sgt[threadIdx.x] = gtseg[(size_t)b * Gg * 2 + threadIdx.x];
    __syncthreads();

    int l = lvl_of(n);
    // scale[l] = 3*fps*2^l ; mids = sqrt(scale[l]*scale[l+1]) = scale[l]*sqrt(2)
    float s0 = 3.0f * fps[b];
    const float SQ2 = 1.41421356237309515f;
    float lo = (l == 0) ? 0.0f : s0 * (float)(1 << (l - 1)) * SQ2;
    float hi = (l == 5) ? __builtin_inff() : s0 * (float)(1 << l) * SQ2;

    float gr = grids[idx];
    float p1 = pseg[(size_t)idx * 3 + 1];
    float p2 = pseg[(size_t)idx * 3 + 2];
    float ps_ = gr - __expf(p1);
    float pe_ = gr + __expf(p2);
    float plen = pe_ - ps_;

    float seg_sum = 0.0f, iou_sum = 0.0f, cnt = 0.0f;
    unsigned int m32 = 0u;
    #pragma unroll 4
    for (int g = 0; g < Gg; ++g) {
        float ss = sgt[g * 2 + 0];
        float ee = sgt[g * 2 + 1];
        float len = ee - ss;
        bool pos = (len >= lo) & (len < hi) & (gr > ss) & (gr < ee);
        if (pos) {
            m32 |= (1u << g);
            cnt += 1.0f;
            // smooth-L1 on log offsets (finite because gr in (ss,ee))
            float d1 = p1 - __logf(gr - ss);
            float d2 = p2 - __logf(ee - gr);
            float a1 = fabsf(d1), a2 = fabsf(d2);
            seg_sum += (a1 < 1.0f ? 0.5f * d1 * d1 : a1 - 0.5f)
                     + (a2 < 1.0f ? 0.5f * d2 * d2 : a2 - 0.5f);
            // 1 - IoU
            float inter = fmaxf(fminf(pe_, ee) - fmaxf(ps_, ss), 0.0f);
            float uni = plen + len - inter;
            iou_sum += 1.0f - inter / uni;
        }
    }
    pos_mask[idx] = m32;

    __shared__ float sred[3];
    if (threadIdx.x == 0) { sred[0] = 0.f; sred[1] = 0.f; sred[2] = 0.f; }
    __syncthreads();
    atomicAdd(&sred[0], seg_sum);
    atomicAdd(&sred[1], iou_sum);
    atomicAdd(&sred[2], cnt);
    __syncthreads();
    if (threadIdx.x == 0) {
        atomicAdd(&acc[1], sred[0]);
        atomicAdd(&acc[2], sred[1]);
        atomicAdd(&acc[3], sred[2]);
    }
}

// Fused: target labels via V_WMMA_F32_16X16X32_F16 (K = G = 32 exactly; 0/1
// operands -> exact), thresholded and consumed by the focal loss streaming
// over pred_logits (nontemporal: 103 MB touched once).
__global__ void k_focal(const float* __restrict__ logits,
                        const int* __restrict__ labels,
                        const unsigned int* __restrict__ pos_mask,
                        const unsigned char* __restrict__ maskv,
                        float* __restrict__ acc) {
    __shared__ int slab[Gg];
    int b = blockIdx.y;
    if (threadIdx.x < Gg) slab[threadIdx.x] = labels[b * Gg + threadIdx.x];
    __syncthreads();

    int wave = threadIdx.x >> 5;
    int lane = threadIdx.x & 31;
    int tile = blockIdx.x * (blockDim.x >> 5) + wave;  // always in range
    int n0   = tile * 16;
    int row  = lane & 15;
    int hiL  = lane >> 4;

    // Per-lane K-map: lanes 0-15 hold K 0..7 / 16..23, lanes 16-31 hold
    // K 8..15 / 24..31 (ISA 16-bit A/B 16x32 layout). Hoist label values
    // out of the class-tile loop -> branch-free B construction.
    int labk[16];
    #pragma unroll
    for (int i = 0; i < 16; ++i) {
        int k = ((i < 8) ? i : i + 8) + hiL * 8;
        labk[i] = slab[k];
    }

    // A-tile: 16 anchors x 32 GT positive flags from the bitmasks.
    unsigned int m32 = pos_mask[(size_t)b * Nn + n0 + row];
    v16h a;
    #pragma unroll
    for (int i = 0; i < 16; ++i) {
        int k = ((i < 8) ? i : i + 8) + hiL * 8;
        a[i] = ((m32 >> k) & 1u) ? (_Float16)1.0f : (_Float16)0.0f;
    }

    // Valid-mask factors for the 8 anchors this lane's D elements cover
    // (ct-invariant -> hoisted).
    float vz[8];
    #pragma unroll
    for (int r = 0; r < 8; ++r)
        vz[r] = maskv[(size_t)b * Nn + n0 + hiL * 8 + r] ? 0.0f : 1.0f;

    float lsum = 0.0f;
    for (int ct = 0; ct < CTILES; ++ct) {
        int c = ct * 16 + row;  // this lane's class column
        // B-tile: one-hot(labels) for 16 classes. No c<Cc guard needed:
        // labels < 200, so padded columns 200..207 are naturally zero.
        v16h bm;
        #pragma unroll
        for (int i = 0; i < 16; ++i)
            bm[i] = (labk[i] == c) ? (_Float16)1.0f : (_Float16)0.0f;

        v8f d = {0.f, 0.f, 0.f, 0.f, 0.f, 0.f, 0.f, 0.f};
        d = __builtin_amdgcn_wmma_f32_16x16x32_f16(
                false, a, false, bm, (short)0, d, false, false);

        if (c < Cc) {
            // Issue all 8 strided loads first, then compute.
            const float* base = &logits[((size_t)b * Nn + n0 + hiL * 8) * Cc + c];
            float xr[8];
            #pragma unroll
            for (int r = 0; r < 8; ++r)
                xr[r] = __builtin_nontemporal_load(base + (size_t)r * Cc);

            #pragma unroll
            for (int r = 0; r < 8; ++r) {
                float t = (d[r] > 0.5f) ? 1.0f : 0.0f;
                // z = (1-2t)*x ; alpha = 0.75 - 0.5t
                float z = (1.0f - 2.0f * t) * xr[r];
                // q = sigmoid(-z); softplus(z) = -log(q); sigmoid(z) = 1-q
                float q   = 1.0f / (1.0f + __expf(z));
                float sp  = -__logf(q);
                float sig = 1.0f - q;
                float alpha = 0.75f - 0.5f * t;
                lsum += alpha * sp * sig * sig * vz[r];
            }
        }
    }

    __shared__ float sred;
    if (threadIdx.x == 0) sred = 0.0f;
    __syncthreads();
    atomicAdd(&sred, lsum);
    __syncthreads();
    if (threadIdx.x == 0) atomicAdd(&acc[0], sred);
}

__global__ void k_fin(const float* __restrict__ acc, float* __restrict__ out) {
    if (threadIdx.x == 0) {
        float np = fmaxf(acc[3], 1.0f);
        out[0] = acc[0] / np;  // loss_ce
        out[1] = acc[1] / np;  // loss_seg
        out[2] = acc[2] / np;  // loss_iou
    }
}

extern "C" void kernel_launch(void* const* d_in, const int* in_sizes, int n_in,
                              void* d_out, int out_size, void* d_ws, size_t ws_size,
                              hipStream_t stream) {
    const float*         pred_logits   = (const float*)d_in[0];
    const float*         pred_segments = (const float*)d_in[1];
    const float*         grids         = (const float*)d_in[2];
    const float*         fps           = (const float*)d_in[3];
    const float*         gt_segments   = (const float*)d_in[4];
    const int*           gt_labels     = (const int*)d_in[5];
    const unsigned char* mask          = (const unsigned char*)d_in[6];
    float* out = (float*)d_out;

    float*        acc      = (float*)d_ws;
    unsigned int* pos_mask = (unsigned int*)((char*)d_ws + 256);

    k_init<<<1, 32, 0, stream>>>(acc);

    int total = Bn * Nn;  // 129024, multiple of 256
    k_pairs<<<total / 256, 256, 0, stream>>>(
        pred_segments, grids, fps, gt_segments, pos_mask, acc);

    // 8 waves/block, one 16-anchor tile per wave; NTILES = 2016 = 252*8 exact.
    dim3 g2(NTILES / 8, Bn);
    k_focal<<<g2, 256, 0, stream>>>(pred_logits, gt_labels, pos_mask, mask, acc);

    k_fin<<<1, 32, 0, stream>>>(acc, out);
}
